// HOPE_68856915689571
// MI455X (gfx1250) — matrix operand order repaired and checked by reading, not compiled, not run on hardware
//
#include <hip/hip_runtime.h>
#include <math.h>

typedef __attribute__((ext_vector_type(16))) _Float16 v16h;
typedef __attribute__((ext_vector_type(8)))  float    v8f;

constexpr int DIMc   = 512;
constexpr int SEQc   = 1024;
constexpr int Bc     = 2;
constexpr int Hc     = 8;
constexpr int HIDc   = 2048;
constexpr int LAYERSc= 2;
constexpr int LVLc   = 3;
constexpr int VOCABc = 50257;
constexpr int VPADc  = 50304;          // ceil(50257/128)*128
constexpr int Mrows  = Bc * SEQc;      // 2048
constexpr int CHUNK  = 8;              // scan LDS chunk (timesteps)

// ---------------------------------------------------------------------------
// A-fragment (16x32 f16, wave32) swizzled address for element (m, k).
// Layout per tile (512 halves): half index = lane*16 + slot,
//   lane = (m&15) + 16*g ;  k%32 -> (g,slot):
//   [0,8): g=0 slot=k ; [8,16): g=1 slot=k-8 ; [16,24): g=0 slot=k-8 ; [24,32): g=1 slot=k-16
// Tiles ordered (mTile * Ktiles + kTile).
// ---------------------------------------------------------------------------
__device__ __forceinline__ size_t aswz_idx(int m, int k, int Ktiles) {
  int kk = k & 31;
  int g, slot;
  if      (kk < 8)  { g = 0; slot = kk;      }
  else if (kk < 16) { g = 1; slot = kk - 8;  }
  else if (kk < 24) { g = 0; slot = kk - 8;  }
  else              { g = 1; slot = kk - 16; }
  int lane = (m & 15) + 16 * g;
  return ((size_t)(m >> 4) * Ktiles + (k >> 5)) * 512 + lane * 16 + slot;
}

__device__ __forceinline__ float gelu_f(float x) {
  float inner = 0.7978845608028654f * (x + 0.044715f * x * x * x);
  return 0.5f * x * (1.0f + tanhf(inner));
}

// ---------------------------------------------------------------------------
// Weight swizzle: fp32 [K,N] row-major (or transposed source) ->
// f16 B-fragment tiles (32k x 16n), tile order (nTile*Ktiles + kTile),
// within tile: half index = lane*16 + slot, k = 8*(lane>=16) + (slot&7) + 16*(slot>=8), n = lane&15.
// transpose!=0: src is [srcRows, K]; element(k,n) = src[n*K+k], zero if n>=srcRows.
// ---------------------------------------------------------------------------
__global__ __launch_bounds__(256) void swizzleB_kernel(
    const float* __restrict__ src, _Float16* __restrict__ dst,
    int K, int N, int Ktiles, int srcRows, int transpose) {
  int tile = blockIdx.x;                 // = nt*Ktiles + kt
  int nt = tile / Ktiles, kt = tile % Ktiles;
  size_t base = (size_t)tile * 512;
  for (int f = threadIdx.x; f < 512; f += 256) {
    int lane = f >> 4, slot = f & 15;
    int k = kt * 32 + (slot & 7) + ((slot >> 3) ? 16 : 0) + ((lane >> 4) ? 8 : 0);
    int n = nt * 16 + (lane & 15);
    float v = 0.0f;
    if (transpose) { if (n < srcRows) v = src[(size_t)n * K + k]; }
    else           { v = src[(size_t)k * N + n]; }
    dst[base + f] = (_Float16)v;
  }
}

// ---------------------------------------------------------------------------
// Embedding: x[r, :] = tok_emb[ids[r]] + pos_emb[r % S]
// ---------------------------------------------------------------------------
__global__ __launch_bounds__(256) void embed_kernel(
    const int* __restrict__ ids, const float* __restrict__ tok,
    const float* __restrict__ pos, float* __restrict__ x) {
  int r = blockIdx.x;
  int t = ids[r];
  int s = r & (SEQc - 1);
  const float* tr = tok + (size_t)t * DIMc;
  const float* pr = pos + (size_t)s * DIMc;
  float* xr = x + (size_t)r * DIMc;
  for (int c = threadIdx.x; c < DIMc; c += 256) xr[c] = tr[c] + pr[c];
}

// ---------------------------------------------------------------------------
// LayerNorm one row per wave32, output directly as A-fragment f16.
// ---------------------------------------------------------------------------
__global__ __launch_bounds__(256) void ln_swz_kernel(
    const float* __restrict__ x, const float* __restrict__ g,
    const float* __restrict__ bsh, _Float16* __restrict__ oswz) {
  int wave = threadIdx.x >> 5, lane = threadIdx.x & 31;
  int r = blockIdx.x * 8 + wave;
  const float* xr = x + (size_t)r * DIMc;
  float vals[16];
  float s = 0.0f, ss = 0.0f;
#pragma unroll
  for (int j = 0; j < 16; ++j) {
    float v = xr[lane * 16 + j];
    vals[j] = v; s += v; ss += v * v;
  }
#pragma unroll
  for (int o = 16; o > 0; o >>= 1) {
    s  += __shfl_xor(s,  o, 32);
    ss += __shfl_xor(ss, o, 32);
  }
  float mu  = s * (1.0f / DIMc);
  float var = ss * (1.0f / DIMc) - mu * mu;
  float rstd = rsqrtf(var + 1e-5f);
#pragma unroll
  for (int j = 0; j < 16; ++j) {
    int n = lane * 16 + j;
    float v = (vals[j] - mu) * rstd * g[n] + bsh[n];
    oswz[aswz_idx(r, n, DIMc / 32)] = (_Float16)v;
  }
}

// ---------------------------------------------------------------------------
// Fast-weight recurrence: one block per (b,h). M[64][64] fp32 in registers
// (16/thread); q/k/v streamed through double-buffered LDS in 8-step chunks.
// M_t = M_{t-1} + v_t k_t^T ; y_t = M_t q_t. y written as A-fragment f16.
// ---------------------------------------------------------------------------
__global__ __launch_bounds__(256) void scan_kernel(
    const float* __restrict__ qf, const float* __restrict__ kf,
    const float* __restrict__ vf, _Float16* __restrict__ ysz) {
  __shared__ float buf[2][3][CHUNK * 64];     // k,q,v  (12 KB)
  int bh = blockIdx.x;
  int b = bh >> 3, h = bh & 7;
  int tid = threadIdx.x;
  int d  = tid >> 2;                 // output/state row (0..63)
  int kb = (tid & 3) * 16;           // state column block (16 cols)
  size_t rowBase = (size_t)b * SEQc;
  int colBase = h * 64;

  float M[16];
#pragma unroll
  for (int j = 0; j < 16; ++j) M[j] = 0.0f;

  auto loadChunk = [&](int c, int nb) {
    int s0 = c * CHUNK;
    for (int e = tid; e < 3 * CHUNK * 64; e += 256) {
      int tsr = e / (CHUNK * 64);
      int w   = e % (CHUNK * 64);
      int u = w >> 6, dd = w & 63;
      const float* src = (tsr == 0) ? kf : ((tsr == 1) ? qf : vf);
      buf[nb][tsr][w] = src[(rowBase + s0 + u) * DIMc + colBase + dd];
    }
  };

  loadChunk(0, 0);
  __syncthreads();
  const int NC = SEQc / CHUNK;
  for (int c = 0; c < NC; ++c) {
    int nb = c & 1;
    if (c + 1 < NC) loadChunk(c + 1, nb ^ 1);
    for (int u = 0; u < CHUNK; ++u) {
      float vd = buf[nb][2][u * 64 + d];
      float yp = 0.0f;
#pragma unroll
      for (int j = 0; j < 16; ++j) {
        float kj = buf[nb][0][u * 64 + kb + j];
        float qj = buf[nb][1][u * 64 + kb + j];
        M[j] = fmaf(vd, kj, M[j]);
        yp   = fmaf(M[j], qj, yp);
      }
      yp += __shfl_xor(yp, 1, 32);
      yp += __shfl_xor(yp, 2, 32);
      if ((tid & 3) == 0) {
        int r = (int)rowBase + c * CHUNK + u;
        ysz[aswz_idx(r, colBase + d, DIMc / 32)] = (_Float16)yp;
      }
    }
    __syncthreads();
  }
}

// ---------------------------------------------------------------------------
// WMMA f16 GEMM: C = A @ B (+bias)(gelu)(+res). A/B pre-swizzled fragment-major.
// Block = 256 thr = 8 waves (2x4), 128x128 tile; wave = 64x32 = 4x2 WMMA tiles.
// Fragment pointers march by a constant 1 KB per K-tile (no per-iteration
// address recomputation); next tile prefetched with regular-temporal hint.
// ---------------------------------------------------------------------------
template <bool BIAS, bool GELU, bool RES, bool WF32, bool WSWZ>
__global__ __launch_bounds__(256) void gemm_wmma_kernel(
    const _Float16* __restrict__ A, const _Float16* __restrict__ Bm,
    const float* __restrict__ bias, const float* res,
    float* outf, _Float16* oswz,
    int N, int ldc, int Ktiles, int KtilesOut) {
  int lane = threadIdx.x & 31;
  int wave = threadIdx.x >> 5;
  int waveM = wave >> 2, waveN = wave & 3;
  int mBase = blockIdx.y * 128 + waveM * 64;
  int nBase = blockIdx.x * 128 + waveN * 32;
  int mt0 = mBase >> 4, nt0 = nBase >> 4;

  // Marching fragment pointers: per-kt stride = 512 halves (1 KB), constant.
  const _Float16* aPtr0 = A  + ((size_t)(mt0 + 0) * Ktiles) * 512 + lane * 16;
  const _Float16* aPtr1 = A  + ((size_t)(mt0 + 1) * Ktiles) * 512 + lane * 16;
  const _Float16* aPtr2 = A  + ((size_t)(mt0 + 2) * Ktiles) * 512 + lane * 16;
  const _Float16* aPtr3 = A  + ((size_t)(mt0 + 3) * Ktiles) * 512 + lane * 16;
  const _Float16* bPtr0 = Bm + ((size_t)(nt0 + 0) * Ktiles) * 512 + lane * 16;
  const _Float16* bPtr1 = Bm + ((size_t)(nt0 + 1) * Ktiles) * 512 + lane * 16;

  v8f acc[4][2];
#pragma unroll
  for (int i = 0; i < 4; ++i)
#pragma unroll
    for (int j = 0; j < 2; ++j)
#pragma unroll
      for (int r = 0; r < 8; ++r) acc[i][j][r] = 0.0f;

  for (int kt = 0; kt < Ktiles; ++kt) {
    v16h aF[4], bF[2];
    aF[0] = *(const v16h*)aPtr0;
    aF[1] = *(const v16h*)aPtr1;
    aF[2] = *(const v16h*)aPtr2;
    aF[3] = *(const v16h*)aPtr3;
    bF[0] = *(const v16h*)bPtr0;
    bF[1] = *(const v16h*)bPtr1;
    // regular-temporal prefetch of the next K-tile (weights are L2-resident)
    __builtin_prefetch(aPtr0 + 512, 0, 3);
    __builtin_prefetch(bPtr0 + 512, 0, 3);
    aPtr0 += 512; aPtr1 += 512; aPtr2 += 512; aPtr3 += 512;
    bPtr0 += 512; bPtr1 += 512;
#pragma unroll
    for (int i = 0; i < 4; ++i)
#pragma unroll
      for (int j = 0; j < 2; ++j)
        acc[i][j] = __builtin_amdgcn_wmma_f32_16x16x32_f16(
            false, aF[i], false, bF[j], (short)0, acc[i][j], false, false);
  }

  int lhi = lane >> 4, llo = lane & 15;
#pragma unroll
  for (int i = 0; i < 4; ++i) {
#pragma unroll
    for (int j = 0; j < 2; ++j) {
      int n = nBase + j * 16 + llo;
      float bv = 0.0f;
      if (BIAS) bv = bias[n];
#pragma unroll
      for (int r = 0; r < 8; ++r) {
        int m = mBase + i * 16 + 8 * lhi + r;
        float val = acc[i][j][r];
        if (BIAS) val += bv;
        if (GELU) val = gelu_f(val);
        if (RES)  val += res[(size_t)m * ldc + n];
        if (WF32) { if (n < N) outf[(size_t)m * ldc + n] = val; }
        if (WSWZ) oswz[aswz_idx(m, n, KtilesOut)] = (_Float16)val;
      }
    }
  }
}

// ---------------------------------------------------------------------------
extern "C" void kernel_launch(void* const* d_in, const int* in_sizes, int n_in,
                              void* d_out, int out_size, void* d_ws, size_t ws_size,
                              hipStream_t stream) {
  (void)in_sizes; (void)n_in; (void)out_size; (void)ws_size;
  const int*   ids   = (const int*)  d_in[0];
  const float* tok   = (const float*)d_in[1];
  const float* pos   = (const float*)d_in[2];
  const float* Wq    = (const float*)d_in[3];
  const float* Wk    = (const float*)d_in[4];
  const float* Wv    = (const float*)d_in[5];
  const float* Wo    = (const float*)d_in[6];
  const float* ln1g  = (const float*)d_in[7];
  const float* ln1b  = (const float*)d_in[8];
  const float* ln2g  = (const float*)d_in[9];
  const float* ln2b  = (const float*)d_in[10];
  const float* lnfg  = (const float*)d_in[11];
  const float* lnfb  = (const float*)d_in[12];
  const float* W1    = (const float*)d_in[13];
  const float* b1    = (const float*)d_in[14];
  const float* W2    = (const float*)d_in[15];
  const float* b2    = (const float*)d_in[16];
  float* out = (float*)d_out;

  // ---- workspace carve ----
  char* p = (char*)d_ws;
  auto take = [&](size_t bytes) -> char* {
    char* r = p; p += (bytes + 255) & ~(size_t)255; return r;
  };
  const size_t WSZ = (size_t)DIMc * DIMc;                 // halves per attn mat
  _Float16* wAttn = (_Float16*)take((size_t)LAYERSc * 4 * WSZ * 2);
  _Float16* w1s   = (_Float16*)take((size_t)LAYERSc * LVLc * DIMc * HIDc * 2);
  _Float16* w2s   = (_Float16*)take((size_t)LAYERSc * LVLc * HIDc * DIMc * 2);
  _Float16* heads = (_Float16*)take((size_t)DIMc * VPADc * 2);
  float*    x     = (float*)   take((size_t)Mrows * DIMc * 4);
  _Float16* xn    = (_Float16*)take((size_t)Mrows * DIMc * 2);
  float*    qb    = (float*)   take((size_t)Mrows * DIMc * 4);
  float*    kbuf  = (float*)   take((size_t)Mrows * DIMc * 4);
  float*    vbuf  = (float*)   take((size_t)Mrows * DIMc * 4);
  _Float16* ysz   = (_Float16*)take((size_t)Mrows * DIMc * 2);
  _Float16* hswz  = (_Float16*)take((size_t)Mrows * HIDc * 2);
  _Float16* cswz  = (_Float16*)take((size_t)Mrows * DIMc * 2);

  // ---- one-shot (per call) weight swizzles to f16 fragment layout ----
  for (int l = 0; l < LAYERSc; ++l) {
    const float* ws4[4] = { Wq + (size_t)l * WSZ, Wk + (size_t)l * WSZ,
                            Wv + (size_t)l * WSZ, Wo + (size_t)l * WSZ };
    for (int w = 0; w < 4; ++w)
      swizzleB_kernel<<<32 * 16, 256, 0, stream>>>(
          ws4[w], wAttn + (size_t)(l * 4 + w) * WSZ, DIMc, DIMc, 16, 0, 0);
  }
  for (int lm = 0; lm < LAYERSc * LVLc; ++lm) {
    swizzleB_kernel<<<128 * 16, 256, 0, stream>>>(
        W1 + (size_t)lm * DIMc * HIDc, w1s + (size_t)lm * DIMc * HIDc,
        DIMc, HIDc, 16, 0, 0);
    swizzleB_kernel<<<32 * 64, 256, 0, stream>>>(
        W2 + (size_t)lm * HIDc * DIMc, w2s + (size_t)lm * HIDc * DIMc,
        HIDc, DIMc, 64, 0, 0);
  }
  swizzleB_kernel<<<(VPADc / 16) * 16, 256, 0, stream>>>(
      tok, heads, DIMc, VPADc, 16, VOCABc, 1);   // transposed tied head

  // ---- embedding ----
  embed_kernel<<<Mrows, 256, 0, stream>>>(ids, tok, pos, x);

  // ---- transformer layers ----
  for (int l = 0; l < LAYERSc; ++l) {
    // LN1 -> xn (f16 A-frag)
    ln_swz_kernel<<<Mrows / 8, 256, 0, stream>>>(x, ln1g + l * DIMc, ln1b + l * DIMc, xn);
    // q,k,v projections (fp32 out)
    gemm_wmma_kernel<false, false, false, true, false><<<dim3(4, 16), 256, 0, stream>>>(
        xn, wAttn + (size_t)(l * 4 + 0) * WSZ, nullptr, nullptr, qb, nullptr,
        DIMc, DIMc, DIMc / 32, 0);
    gemm_wmma_kernel<false, false, false, true, false><<<dim3(4, 16), 256, 0, stream>>>(
        xn, wAttn + (size_t)(l * 4 + 1) * WSZ, nullptr, nullptr, kbuf, nullptr,
        DIMc, DIMc, DIMc / 32, 0);
    gemm_wmma_kernel<false, false, false, true, false><<<dim3(4, 16), 256, 0, stream>>>(
        xn, wAttn + (size_t)(l * 4 + 2) * WSZ, nullptr, nullptr, vbuf, nullptr,
        DIMc, DIMc, DIMc / 32, 0);
    // fast-weight recurrence -> ysz (f16 A-frag)
    scan_kernel<<<Bc * Hc, 256, 0, stream>>>(qb, kbuf, vbuf, ysz);
    // output proj + residual into x
    gemm_wmma_kernel<false, false, true, true, false><<<dim3(4, 16), 256, 0, stream>>>(
        ysz, wAttn + (size_t)(l * 4 + 3) * WSZ, nullptr, x, x, nullptr,
        DIMc, DIMc, DIMc / 32, 0);
    // LN2 -> xn
    ln_swz_kernel<<<Mrows / 8, 256, 0, stream>>>(x, ln2g + l * DIMc, ln2b + l * DIMc, xn);
    // CMS: 3 chained MLP levels
    const _Float16* cin = xn;
    for (int m = 0; m < LVLc; ++m) {
      int lm = l * LVLc + m;
      // h = gelu(cin @ W1 + b1)  -> hswz (f16 A-frag)
      gemm_wmma_kernel<true, true, false, false, true><<<dim3(16, 16), 256, 0, stream>>>(
          cin, w1s + (size_t)lm * DIMc * HIDc, b1 + (size_t)lm * HIDc, nullptr,
          nullptr, hswz, HIDc, HIDc, DIMc / 32, HIDc / 32);
      if (m < LVLc - 1) {
        // c = h @ W2 + b2 -> cswz (feeds next level)
        gemm_wmma_kernel<true, false, false, false, true><<<dim3(4, 16), 256, 0, stream>>>(
            hswz, w2s + (size_t)lm * HIDc * DIMc, b2 + (size_t)lm * DIMc, nullptr,
            nullptr, cswz, DIMc, DIMc, HIDc / 32, DIMc / 32);
        cin = cswz;
      } else {
        // last level: x = x + (h @ W2 + b2)
        gemm_wmma_kernel<true, false, true, true, false><<<dim3(4, 16), 256, 0, stream>>>(
            hswz, w2s + (size_t)lm * HIDc * DIMc, b2 + (size_t)lm * DIMc, x,
            x, nullptr, DIMc, DIMc, HIDc / 32, 0);
      }
    }
  }

  // ---- final LN + tied-head logits ----
  ln_swz_kernel<<<Mrows / 8, 256, 0, stream>>>(x, lnfg, lnfb, xn);
  gemm_wmma_kernel<false, false, false, true, false><<<dim3(VPADc / 128, 16), 256, 0, stream>>>(
      xn, heads, nullptr, nullptr, out, nullptr, VOCABc, VOCABc, DIMc / 32, 0);
}
